// MultiheadMaskedAttention_25709674234354
// MI455X (gfx1250) — matrix operand order repaired and checked by reading, not compiled, not run on hardware
//
#include <hip/hip_runtime.h>
#include <hip/hip_bf16.h>

#define B_   2
#define S_   2048
#define H_   1024
#define NH_  16
#define D_   64
#define H3_  3072
#define MROWS (B_*S_)   // 4096
#define LDK  40         // padded LDS row stride (elements)
#define BUFE (128*LDK)  // elements per LDS tile buffer

typedef __attribute__((ext_vector_type(16))) __bf16       bf16x16;
typedef __attribute__((ext_vector_type(8)))  float        f32x8;
typedef __attribute__((ext_vector_type(4)))  unsigned int u32x4;

#if defined(__has_builtin)
#if __has_builtin(__builtin_amdgcn_global_load_async_to_lds_b128)
#define HAVE_ASYNC_LDS 1
#endif
#endif
#ifndef HAVE_ASYNC_LDS
#define HAVE_ASYNC_LDS 0
#endif

union FragU { bf16x16 v; u32x4 q[2]; };

// A-operand fragment: two 16B chunks (callers use contiguous K: lane<16: K=0..15, lane>=16: K=16..31)
__device__ __forceinline__ bf16x16 frag_2x16(const __bf16* p0, const __bf16* p1) {
  FragU f;
  f.q[0] = *(const u32x4*)p0;
  f.q[1] = *(const u32x4*)p1;
  return f.v;
}
// B-operand fragment: one contiguous 32B chunk (lane<16: K=0..15, lane>=16: K=16..31)
__device__ __forceinline__ bf16x16 frag_32(const __bf16* p) {
  FragU f;
  f.q[0] = ((const u32x4*)p)[0];
  f.q[1] = ((const u32x4*)p)[1];
  return f.v;
}
__device__ __forceinline__ f32x8 wmma_bf16(bf16x16 a, bf16x16 b, f32x8 c) {
  return __builtin_amdgcn_wmma_f32_16x16x32_bf16(false, a, false, b, (short)0, c,
                                                 false, false);
}

#if HAVE_ASYNC_LDS
// builtin signature (from hipcc diagnostic): param 1 is
//   `__attribute__((vector_size(16))) int __device__ *`  (global int4*, non-const)
typedef int v4i_gcc __attribute__((vector_size(16)));
typedef __attribute__((address_space(1))) v4i_gcc* gp128_t;
typedef __attribute__((address_space(3))) v4i_gcc* lp128_t;

__device__ __forceinline__ void async_cp16(const void* g, void* l) {
  __builtin_amdgcn_global_load_async_to_lds_b128(
      (gp128_t)(unsigned long long)g,
      (lp128_t)(unsigned int)(unsigned long long)l, 0, 0);
}
__device__ __forceinline__ void wait_async0() {
#if __has_builtin(__builtin_amdgcn_s_wait_asynccnt)
  __builtin_amdgcn_s_wait_asynccnt(0);
#else
  asm volatile("s_wait_asynccnt 0" ::: "memory");
#endif
}
#endif

// ---------------- precision conversion kernels ----------------
__global__ void cvt_f32_bf16(const float* __restrict__ src, __bf16* __restrict__ dst, int n) {
  int i = blockIdx.x * blockDim.x + threadIdx.x;
  if (i < n) dst[i] = (__bf16)src[i];
}
// dst[c][r] = src[r][c] : store weights N-major so WMMA B-fragments are contiguous
__global__ void cvt_transpose(const float* __restrict__ src, __bf16* __restrict__ dst,
                              int rows, int cols) {
  int i = blockIdx.x * blockDim.x + threadIdx.x;
  if (i < rows * cols) {
    int r = i / cols, c = i - r * cols;
    dst[(size_t)c * rows + r] = (__bf16)src[i];
  }
}

// ---------------- staging helpers (2 chunks of 16B per thread per matrix) ----------------
struct Stage { u32x4 a0, a1, b0, b1; };

__device__ __forceinline__ void stage_load(const __bf16* __restrict__ A,
                                           const __bf16* __restrict__ Bm,
                                           int Kdim, int m0, int n0, int kb, int t,
                                           Stage& s) {
  const int row = t >> 2, col = (t & 3) << 3;
  s.a0 = *(const u32x4*)&A [(size_t)(m0 + row)      * Kdim + kb + col];
  s.b0 = *(const u32x4*)&Bm[(size_t)(n0 + row)      * Kdim + kb + col];
  s.a1 = *(const u32x4*)&A [(size_t)(m0 + 64 + row) * Kdim + kb + col];
  s.b1 = *(const u32x4*)&Bm[(size_t)(n0 + 64 + row) * Kdim + kb + col];
}
__device__ __forceinline__ void stage_store(__bf16* smA, __bf16* smB, int t,
                                            const Stage& s) {
  const int row = t >> 2, col = (t & 3) << 3;
  *(u32x4*)&smA[row * LDK + col]        = s.a0;
  *(u32x4*)&smB[row * LDK + col]        = s.b0;
  *(u32x4*)&smA[(64 + row) * LDK + col] = s.a1;
  *(u32x4*)&smB[(64 + row) * LDK + col] = s.b1;
}
#if HAVE_ASYNC_LDS
__device__ __forceinline__ void stage_async(const __bf16* __restrict__ A,
                                            const __bf16* __restrict__ Bm,
                                            int Kdim, int m0, int n0, int kb, int t,
                                            __bf16* smA, __bf16* smB) {
  const int row = t >> 2, col = (t & 3) << 3;
  async_cp16(&A [(size_t)(m0 + row)      * Kdim + kb + col], &smA[row * LDK + col]);
  async_cp16(&Bm[(size_t)(n0 + row)      * Kdim + kb + col], &smB[row * LDK + col]);
  async_cp16(&A [(size_t)(m0 + 64 + row) * Kdim + kb + col], &smA[(64 + row) * LDK + col]);
  async_cp16(&Bm[(size_t)(n0 + 64 + row) * Kdim + kb + col], &smB[(64 + row) * LDK + col]);
}
#endif

// ---------------- shared WMMA GEMM mainloop (double-buffered, 1 barrier/step) --------
// C[128x128] tile: 8 waves, wave = 32(M) x 64(N); A:[M][K] bf16, Bm:[N][K] bf16 (transposed)
__device__ __forceinline__ void gemm_tile(const __bf16* __restrict__ A,
                                          const __bf16* __restrict__ Bm,
                                          int Kdim, int m0, int n0,
                                          __bf16* smA, __bf16* smB,
                                          f32x8 acc[2][4]) {
  const int t    = threadIdx.x;
  const int lane = t & 31, w = t >> 5;
  const int lr   = lane & 15, hi = lane >> 4;
  const int wm   = (w & 3) * 32;
  const int wn   = (w >> 2) * 64;

#if HAVE_ASYNC_LDS
  stage_async(A, Bm, Kdim, m0, n0, 0, t, smA, smB);
  wait_async0();
  __syncthreads();
#else
  Stage s;
  stage_load(A, Bm, Kdim, m0, n0, 0, t, s);
  stage_store(smA, smB, t, s);
  __syncthreads();
#endif

  for (int kb = 0; kb < Kdim; kb += 32) {
    const int  cur  = (kb >> 5) & 1;
    const bool more = (kb + 32) < Kdim;
    __bf16* cA = smA + cur * BUFE;
    __bf16* cB = smB + cur * BUFE;
    __bf16* nA = smA + (1 - cur) * BUFE;
    __bf16* nB = smB + (1 - cur) * BUFE;

#if HAVE_ASYNC_LDS
    if (more) stage_async(A, Bm, Kdim, m0, n0, kb + 32, t, nA, nB);
#else
    if (more) stage_load(A, Bm, Kdim, m0, n0, kb + 32, t, s);
#endif

    bf16x16 af[2], bfr[4];
#pragma unroll
    for (int mi = 0; mi < 2; ++mi) {
      const __bf16* rp = &cA[(wm + mi * 16 + lr) * LDK];
      af[mi] = frag_2x16(rp + hi * 16, rp + hi * 16 + 8);
    }
#pragma unroll
    for (int ni = 0; ni < 4; ++ni) {
      const __bf16* rp = &cB[(wn + ni * 16 + lr) * LDK];
      bfr[ni] = frag_32(rp + hi * 16);
    }
#pragma unroll
    for (int mi = 0; mi < 2; ++mi)
#pragma unroll
      for (int ni = 0; ni < 4; ++ni)
        acc[mi][ni] = wmma_bf16(af[mi], bfr[ni], acc[mi][ni]);

#if HAVE_ASYNC_LDS
    if (more) wait_async0();
#else
    if (more) stage_store(nA, nB, t, s);  // s_wait_loadcnt inserted here by compiler
#endif
    __syncthreads();
  }
}

// ---------------- QKV projection ----------------
__global__ __launch_bounds__(256) void qkv_gemm(const __bf16* __restrict__ xb,
                                                const __bf16* __restrict__ wT,
                                                const float* __restrict__ bias,
                                                __bf16* __restrict__ Qb,
                                                __bf16* __restrict__ Kb,
                                                __bf16* __restrict__ Vt) {
  __shared__ __bf16 smA[2 * BUFE];
  __shared__ __bf16 smB[2 * BUFE];
  const int n0 = blockIdx.x * 128, m0 = blockIdx.y * 128;
  f32x8 acc[2][4] = {};
  gemm_tile(xb, wT, H_, m0, n0, smA, smB, acc);

  const int lane = threadIdx.x & 31, w = threadIdx.x >> 5;
  const int lr = lane & 15, hi = lane >> 4;
  const int wm = (w & 3) * 32, wn = (w >> 2) * 64;
#pragma unroll
  for (int mi = 0; mi < 2; ++mi)
#pragma unroll
    for (int ni = 0; ni < 4; ++ni)
#pragma unroll
      for (int r = 0; r < 8; ++r) {
        int m = m0 + wm + mi * 16 + r + 8 * hi;
        int n = n0 + wn + ni * 16 + lr;
        float v = acc[mi][ni][r] + bias[n];
        int which = n >> 10, wi = n & 1023;
        int head = wi >> 6, d = wi & 63;
        int bidx = m >> 11, sdx = m & 2047;
        size_t bh = (size_t)(bidx * NH_ + head);
        if (which == 0)       Qb[(bh * S_ + sdx) * D_ + d] = (__bf16)v;
        else if (which == 1)  Kb[(bh * S_ + sdx) * D_ + d] = (__bf16)v;
        else                  Vt[(bh * D_ + d) * S_ + sdx] = (__bf16)v;  // V transposed
      }
}

// ---------------- flash-style causal attention ----------------
__global__ __launch_bounds__(256) void attn_kernel(const __bf16* __restrict__ Qb,
                                                   const __bf16* __restrict__ Kb,
                                                   const __bf16* __restrict__ Vt,
                                                   __bf16* __restrict__ attnb) {
  __shared__ __bf16 Pl[8 * 16 * LDK];   // per-wave 16x32 probability staging
  const int bh = blockIdx.x;            // b*16 + head
  const int bidx = bh >> 4, head = bh & 15;
  const int t = threadIdx.x, lane = t & 31, w = t >> 5;
  const int lr = lane & 15, hi = lane >> 4;
  const int q0 = blockIdx.y * 128 + w * 16;

  const __bf16* Qp = Qb + (size_t)bh * S_ * D_;
  const __bf16* Kp = Kb + (size_t)bh * S_ * D_;
  const __bf16* Vp = Vt + (size_t)bh * D_ * S_;
  __bf16* Pw = &Pl[w * 16 * LDK];

  bf16x16 qf[2];
  {
    const __bf16* rp = Qp + (size_t)(q0 + lr) * D_;
    qf[0] = frag_2x16(rp + hi * 16,      rp + hi * 16 + 8);
    qf[1] = frag_2x16(rp + 32 + hi * 16, rp + 40 + hi * 16);
  }
  f32x8 oacc[4] = {};
  float mrow[8], lrow[8];
#pragma unroll
  for (int r = 0; r < 8; ++r) { mrow[r] = -3.0e38f; lrow[r] = 0.f; }
  const float LOG2E = 1.4426950408889634f;
  const float scale = 0.125f;   // 1/sqrt(64)
  const int kmax = q0 + 15;

  for (int k0 = 0; k0 <= kmax; k0 += 32) {
    f32x8 st[2] = {};
#pragma unroll
    for (int tt = 0; tt < 2; ++tt) {
      const __bf16* kp = Kp + (size_t)(k0 + tt * 16 + lr) * D_;
      st[tt] = wmma_bf16(qf[0], frag_32(kp + hi * 16),      st[tt]);
      st[tt] = wmma_bf16(qf[1], frag_32(kp + 32 + hi * 16), st[tt]);
    }
#pragma unroll
    for (int r = 0; r < 8; ++r) {
      int qrow = q0 + r + 8 * hi;
      float s0 = (k0 + lr      <= qrow) ? st[0][r] * scale : -3.0e38f;
      float s1 = (k0 + 16 + lr <= qrow) ? st[1][r] * scale : -3.0e38f;
      float v = fmaxf(s0, s1);
      v = fmaxf(v, __shfl_xor(v, 1));
      v = fmaxf(v, __shfl_xor(v, 2));
      v = fmaxf(v, __shfl_xor(v, 4));
      v = fmaxf(v, __shfl_xor(v, 8));
      float mn   = fmaxf(mrow[r], v);
      float corr = exp2f((mrow[r] - mn) * LOG2E);
      float p0   = exp2f((s0 - mn) * LOG2E);
      float p1   = exp2f((s1 - mn) * LOG2E);
      float rs = p0 + p1;
      rs += __shfl_xor(rs, 1);
      rs += __shfl_xor(rs, 2);
      rs += __shfl_xor(rs, 4);
      rs += __shfl_xor(rs, 8);
      lrow[r] = lrow[r] * corr + rs;
      mrow[r] = mn;
#pragma unroll
      for (int dt = 0; dt < 4; ++dt) oacc[dt][r] *= corr;
      int prow = r + 8 * hi;
      Pw[prow * LDK + lr]      = (__bf16)p0;
      Pw[prow * LDK + 16 + lr] = (__bf16)p1;
    }
    asm volatile("s_wait_dscnt 0" ::: "memory");  // LDS transpose round-trip
    bf16x16 pf;
    {
      const __bf16* rp = &Pw[lr * LDK];
      pf = frag_2x16(rp + hi * 16, rp + hi * 16 + 8);
    }
#pragma unroll
    for (int dt = 0; dt < 4; ++dt) {
      const __bf16* vp = Vp + (size_t)(dt * 16 + lr) * S_ + k0;
      oacc[dt] = wmma_bf16(pf, frag_32(vp + hi * 16), oacc[dt]);
    }
  }

  __bf16* op = attnb + (size_t)bidx * S_ * H_;
#pragma unroll
  for (int dt = 0; dt < 4; ++dt)
#pragma unroll
    for (int r = 0; r < 8; ++r) {
      int q = q0 + r + 8 * hi;
      int col = head * 64 + dt * 16 + lr;
      op[(size_t)q * H_ + col] = (__bf16)(oacc[dt][r] * (1.0f / lrow[r]));
    }
}

// ---------------- output projection ----------------
__global__ __launch_bounds__(256) void out_gemm(const __bf16* __restrict__ attnb,
                                                const __bf16* __restrict__ woT,
                                                const float* __restrict__ bias,
                                                float* __restrict__ out) {
  __shared__ __bf16 smA[2 * BUFE];
  __shared__ __bf16 smB[2 * BUFE];
  const int n0 = blockIdx.x * 128, m0 = blockIdx.y * 128;
  f32x8 acc[2][4] = {};
  gemm_tile(attnb, woT, H_, m0, n0, smA, smB, acc);

  const int lane = threadIdx.x & 31, w = threadIdx.x >> 5;
  const int lr = lane & 15, hi = lane >> 4;
  const int wm = (w & 3) * 32, wn = (w >> 2) * 64;
#pragma unroll
  for (int mi = 0; mi < 2; ++mi)
#pragma unroll
    for (int ni = 0; ni < 4; ++ni)
#pragma unroll
      for (int r = 0; r < 8; ++r) {
        int m = m0 + wm + mi * 16 + r + 8 * hi;
        int n = n0 + wn + ni * 16 + lr;
        out[(size_t)m * H_ + n] = acc[mi][ni][r] + bias[n];
      }
}

// ---------------- launcher ----------------
extern "C" void kernel_launch(void* const* d_in, const int* in_sizes, int n_in,
                              void* d_out, int out_size, void* d_ws, size_t ws_size,
                              hipStream_t stream) {
  const float* x     = (const float*)d_in[0];
  const float* W_qkv = (const float*)d_in[1];
  const float* b_qkv = (const float*)d_in[2];
  const float* W_o   = (const float*)d_in[3];
  const float* b_o   = (const float*)d_in[4];
  float* out = (float*)d_out;

  __bf16* xb    = (__bf16*)d_ws;                  // [4096][1024]
  __bf16* wqkvT = xb    + (size_t)MROWS * H_;     // [3072][1024] (W_qkv^T)
  __bf16* woT   = wqkvT + (size_t)H3_ * H_;       // [1024][1024] (W_o^T)
  __bf16* Qb    = woT   + (size_t)H_ * H_;        // [b][h][s][d]
  __bf16* Kb    = Qb    + (size_t)B_ * NH_ * S_ * D_;
  __bf16* Vt    = Kb    + (size_t)B_ * NH_ * S_ * D_;   // [b][h][d][s]
  __bf16* attnb = Vt    + (size_t)B_ * NH_ * S_ * D_;   // [4096][1024]

  {
    int n = MROWS * H_;
    cvt_f32_bf16<<<(n + 255) / 256, 256, 0, stream>>>(x, xb, n);
  }
  {
    int n = H_ * H3_;
    cvt_transpose<<<(n + 255) / 256, 256, 0, stream>>>(W_qkv, wqkvT, H_, H3_);
  }
  {
    int n = H_ * H_;
    cvt_transpose<<<(n + 255) / 256, 256, 0, stream>>>(W_o, woT, H_, H_);
  }
  {
    dim3 g(H3_ / 128, MROWS / 128);   // 24 x 32
    qkv_gemm<<<g, 256, 0, stream>>>(xb, wqkvT, b_qkv, Qb, Kb, Vt);
  }
  {
    dim3 g(B_ * NH_, S_ / 128);       // 32 x 16
    attn_kernel<<<g, 256, 0, stream>>>(Qb, Kb, Vt, attnb);
  }
  {
    dim3 g(H_ / 128, MROWS / 128);    // 8 x 32
    out_gemm<<<g, 256, 0, stream>>>(attnb, woT, b_o, out);
  }
}
